// S4Model_2465311228333
// MI455X (gfx1250) — matrix operand order repaired and checked by reading, not compiled, not run on hardware
//
#include <hip/hip_runtime.h>
#include <hip/hip_bf16.h>
#include <math.h>
#include <stdint.h>

#define BATCH 16
#define LSEQ  4096
#define HDIM  128
#define NSTATE2 32
#define NLAYER 4
#define TCH   64   // chunk length
#define NCH   64   // number of chunks (LSEQ / TCH)
#define DOUT  100

typedef __bf16 bf16t;
typedef __attribute__((ext_vector_type(16))) __bf16 v16bf;
typedef __attribute__((ext_vector_type(8)))  float  v8f;

// ---------------------------------------------------------------------------
// WMMA helpers (CDNA5 wave32, V_WMMA_F32_16X16X32_BF16)
// ---------------------------------------------------------------------------
__device__ __forceinline__ v8f wmma_bf16(v16bf a, v16bf b, v8f c) {
  return __builtin_amdgcn_wmma_f32_16x16x32_bf16(false, a, false, b,
                                                 (short)0, c, false, false);
}

// A fragment: 16x32 (MxK) bf16, row-major source p[m*ld + k]
// Layout: lanes 0-15 M=lane, K=0..15 over V0..V7;
//         lanes16-31 same M, K=16..31.
__device__ __forceinline__ v16bf load_a_frag(const bf16t* p, int ld) {
  const int lane = threadIdx.x & 31;
  const int m  = lane & 15;
  const int hi = lane >> 4;
  v16bf a;
#pragma unroll
  for (int v = 0; v < 8; ++v) {
    const int k0 = 2 * v + 16 * hi;
    a[2 * v]     = p[m * ld + k0];
    a[2 * v + 1] = p[m * ld + k0 + 1];
  }
  return a;
}

// B fragment: 32x16 (KxN) bf16 from N-major storage p[n*ld + k]
// lanes 0-15: N=lane, K=0..15 over V0..V7; lanes 16-31: K=16..31.
__device__ __forceinline__ v16bf load_b_frag(const bf16t* p, int ld) {
  const int lane = threadIdx.x & 31;
  const int n  = lane & 15;
  const int hi = lane >> 4;
  v16bf b;
#pragma unroll
  for (int v = 0; v < 8; ++v) {
    const int k0 = 2 * v + 16 * hi;
    b[2 * v]     = p[n * ld + k0];
    b[2 * v + 1] = p[n * ld + k0 + 1];
  }
  return b;
}

// Same but f32 source with inline convert to bf16.
__device__ __forceinline__ v16bf load_b_frag_f32(const float* p, int ld) {
  const int lane = threadIdx.x & 31;
  const int n  = lane & 15;
  const int hi = lane >> 4;
  v16bf b;
#pragma unroll
  for (int v = 0; v < 8; ++v) {
    const int k0 = 2 * v + 16 * hi;
    b[2 * v]     = (bf16t)p[n * ld + k0];
    b[2 * v + 1] = (bf16t)p[n * ld + k0 + 1];
  }
  return b;
}

// D fragment store: VGPR r -> row (r + 8*(lane>=16)), col (lane&15).
__device__ __forceinline__ void store_d_frag(float* p, int ld, v8f d) {
  const int lane = threadIdx.x & 31;
  const int n  = lane & 15;
  const int hi = lane >> 4;
#pragma unroll
  for (int r = 0; r < 8; ++r) p[(r + 8 * hi) * ld + n] = d[r];
}

// ---------------------------------------------------------------------------
// Kernel 0: encoder  h[b,hh,l] = x[b,l,0]*enc_w[hh] + enc_b[hh]
// ---------------------------------------------------------------------------
__global__ __launch_bounds__(256) void encode_kernel(
    const float* __restrict__ x, const float* __restrict__ enc_w,
    const float* __restrict__ enc_b, float* __restrict__ hbuf) {
  size_t idx = (size_t)blockIdx.x * 256 + threadIdx.x; // B*H*L
  int l  = (int)(idx & (LSEQ - 1));
  size_t t = idx >> 12;
  int hh = (int)(t & (HDIM - 1));
  int b  = (int)(t >> 7);
  hbuf[idx] = x[(size_t)b * LSEQ + l] * enc_w[hh] + enc_b[hh];
}

// ---------------------------------------------------------------------------
// Kernel 1 (per layer): build per-head matrices M1 (state gather), E (carry
// broadcast), Toeplitz(k), w^T, and bf16 copy of out_w.  1 block, 128 threads.
// ---------------------------------------------------------------------------
__global__ __launch_bounds__(128) void setup_kernel(
    const float* __restrict__ log_dt, const float* __restrict__ Cp,
    const float* __restrict__ log_A_real, const float* __restrict__ A_imag,
    const float* __restrict__ out_w, int layer,
    bf16t* __restrict__ M1, bf16t* __restrict__ E, bf16t* __restrict__ Toep,
    float* __restrict__ wT, bf16t* __restrict__ wbf) {
  const int h = threadIdx.x; // 0..127
  const float dt = expf(log_dt[layer * HDIM + h]);
  float karr[TCH];
#pragma unroll
  for (int m = 0; m < TCH; ++m) karr[m] = 0.f;

  for (int n = 0; n < NSTATE2; ++n) {
    const int idx = (layer * HDIM + h) * NSTATE2 + n;
    const float Ar = -expf(log_A_real[idx]);
    const float Ai = A_imag[idx];
    const float dAr = Ar * dt, dAi = Ai * dt;
    const float er = expf(dAr);
    const float wr = er * cosf(dAi), wi = er * sinf(dAi);
    const float Cre = Cp[idx * 2 + 0], Cim = Cp[idx * 2 + 1];
    // C_disc = C * (exp(dtA)-1)/A
    const float nr = wr - 1.f, ni = wi;
    const float mag = Ar * Ar + Ai * Ai;
    const float qr = (nr * Ar + ni * Ai) / mag;
    const float qi = (ni * Ar - nr * Ai) / mag;
    const float Cdr = Cre * qr - Cim * qi;
    const float Cdi = Cre * qi + Cim * qr;

    float pr = 1.f, pi = 0.f; // w^m
    for (int m = 0; m <= TCH; ++m) {
      if (m < TCH) {
        karr[m] += 2.f * (Cdr * pr - Cdi * pi);
        const int t = TCH - 1 - m; // M1[2n][t] = Re w^(T-1-t)
        M1[((size_t)h * 64 + 2 * n) * 64 + t]     = (bf16t)pr;
        M1[((size_t)h * 64 + 2 * n + 1) * 64 + t] = (bf16t)pi;
      }
      if (m >= 1) { // E[t][2n] with t = m-1 : y_carry[t] = 2Re(Cd w^(t+1) carry)
        const int t = m - 1;
        E[((size_t)h * 64 + t) * 64 + 2 * n]     = (bf16t)( 2.f * (Cdr * pr - Cdi * pi));
        E[((size_t)h * 64 + t) * 64 + 2 * n + 1] = (bf16t)(-2.f * (Cdr * pi + Cdi * pr));
      }
      if (m == TCH) {
        wT[((size_t)h * NSTATE2 + n) * 2 + 0] = pr;
        wT[((size_t)h * NSTATE2 + n) * 2 + 1] = pi;
      }
      const float npr = pr * wr - pi * wi;
      const float npi = pr * wi + pi * wr;
      pr = npr; pi = npi;
    }
  }
  // Lower-triangular Toeplitz of k (rows t, cols s)
  for (int t = 0; t < TCH; ++t)
    for (int s = 0; s < TCH; ++s)
      Toep[((size_t)h * 64 + t) * 64 + s] = (bf16t)((s <= t) ? karr[t - s] : 0.f);
  // bf16 copy of this layer's out_w (256 x 128, row-major, K contiguous)
  const size_t wo = (size_t)layer * 2 * HDIM * HDIM;
  for (int i = 0; i < 256; ++i)
    wbf[(size_t)h * 256 + i] = (bf16t)out_w[wo + (size_t)h * 256 + i];
}

// ---------------------------------------------------------------------------
// Kernel 2 (per layer): S_local(64x64) = M1_h(64x64) @ U(64x64) per (b,h).
// Block = (b,h), 4 waves; wave w owns M-rows [16w,16w+16).
// ---------------------------------------------------------------------------
__global__ __launch_bounds__(128) void phase1_kernel(
    const bf16t* __restrict__ M1, const float* __restrict__ hbuf,
    float* __restrict__ Slocal) {
  const int bh = blockIdx.x;
  const int h  = bh & (HDIM - 1);
  const int w  = threadIdx.x >> 5;
  const bf16t* A = M1 + (size_t)h * 4096;
  const float* U = hbuf + (size_t)bh * LSEQ;   // (t,c): U[c*64+t] -> N-major, K=t pair contiguous
  float* S = Slocal + (size_t)bh * 4096;       // rows 2n, cols c
  const v16bf a0 = load_a_frag(A + (w * 16) * 64 + 0, 64);
  const v16bf a1 = load_a_frag(A + (w * 16) * 64 + 32, 64);
#pragma unroll
  for (int nt = 0; nt < 4; ++nt) {
    const v16bf b0 = load_b_frag_f32(U + (nt * 16) * 64 + 0, 64);
    const v16bf b1 = load_b_frag_f32(U + (nt * 16) * 64 + 32, 64);
    v8f acc = {};
    acc = wmma_bf16(a0, b0, acc);
    acc = wmma_bf16(a1, b1, acc);
    store_d_frag(S + (w * 16) * 64 + nt * 16, 64, acc);
  }
}

// ---------------------------------------------------------------------------
// Kernel 3 (per layer): inter-chunk scan (64 sequential steps, 65536 lanes).
// carry[c] = w^T * carry[c-1] + S_local[c-1];  carry[0] = 0.
// Carry stored bf16 as (b,h)[c][2n interleaved re/im] -> B operand layout.
// ---------------------------------------------------------------------------
__global__ __launch_bounds__(256) void scan_kernel(
    const float* __restrict__ Slocal, const float* __restrict__ wT,
    bf16t* __restrict__ Carry) {
  const int idx = blockIdx.x * 256 + threadIdx.x; // (b*H+h)*32 + n
  const int n  = idx & (NSTATE2 - 1);
  const int bh = idx >> 5;
  const int h  = bh & (HDIM - 1);
  const float wr = wT[((size_t)h * NSTATE2 + n) * 2 + 0];
  const float wi = wT[((size_t)h * NSTATE2 + n) * 2 + 1];
  const float* S = Slocal + (size_t)bh * 4096;
  bf16t* Cy = Carry + (size_t)bh * 4096;
  float cr = 0.f, ci = 0.f;
  for (int c = 0; c < NCH; ++c) {
    Cy[c * 64 + 2 * n]     = (bf16t)cr;
    Cy[c * 64 + 2 * n + 1] = (bf16t)ci;
    const float sr = S[(2 * n) * 64 + c];
    const float si = S[(2 * n + 1) * 64 + c];
    const float t0 = wr * cr - wi * ci + sr;
    const float t1 = wr * ci + wi * cr + si;
    cr = t0; ci = t1;
  }
}

// ---------------------------------------------------------------------------
// Kernel 4 (per layer): Y = Toep_h @ U + E_h @ Carry ; y += D*u ; GELU(erf);
// store transposed to ybf (B,L,H) bf16 for the GLU GEMM.
// ---------------------------------------------------------------------------
__global__ __launch_bounds__(128) void phase3_kernel(
    const bf16t* __restrict__ Toep, const bf16t* __restrict__ E,
    const bf16t* __restrict__ Carry, const float* __restrict__ hbuf,
    const float* __restrict__ Dp, bf16t* __restrict__ ybf, int layer) {
  const int bh = blockIdx.x;
  const int h  = bh & (HDIM - 1);
  const int b  = bh >> 7;
  const int w    = threadIdx.x >> 5;
  const int lane = threadIdx.x & 31;
  const bf16t* Ta = Toep + (size_t)h * 4096;
  const bf16t* Ea = E    + (size_t)h * 4096;
  const bf16t* Cy = Carry + (size_t)bh * 4096;
  const float* U  = hbuf  + (size_t)bh * LSEQ;
  const float Dh = Dp[layer * HDIM + h];
  const v16bf t0 = load_a_frag(Ta + (w * 16) * 64 + 0, 64);
  const v16bf t1 = load_a_frag(Ta + (w * 16) * 64 + 32, 64);
  const v16bf e0 = load_a_frag(Ea + (w * 16) * 64 + 0, 64);
  const v16bf e1 = load_a_frag(Ea + (w * 16) * 64 + 32, 64);
  const int hi = lane >> 4;
  const int nl = lane & 15;
#pragma unroll
  for (int nt = 0; nt < 4; ++nt) {
    const v16bf ub0 = load_b_frag_f32(U + (nt * 16) * 64 + 0, 64);
    const v16bf ub1 = load_b_frag_f32(U + (nt * 16) * 64 + 32, 64);
    const v16bf cb0 = load_b_frag(Cy + (nt * 16) * 64 + 0, 64);
    const v16bf cb1 = load_b_frag(Cy + (nt * 16) * 64 + 32, 64);
    v8f acc = {};
    acc = wmma_bf16(t0, ub0, acc);
    acc = wmma_bf16(t1, ub1, acc);
    acc = wmma_bf16(e0, cb0, acc);
    acc = wmma_bf16(e1, cb1, acc);
#pragma unroll
    for (int r = 0; r < 8; ++r) {
      const int t = w * 16 + r + 8 * hi;
      const int c = nt * 16 + nl;
      const int l = c * 64 + t;
      const float u = U[l];
      const float y = acc[r] + Dh * u;
      const float ge = 0.5f * y * (1.f + erff(y * 0.70710678118654752f));
      ybf[((size_t)b * LSEQ + l) * HDIM + h] = (bf16t)ge;
    }
  }
}

// ---------------------------------------------------------------------------
// Kernel 5 (per layer): z = out_w(256x128) @ y ; GLU + residual + LayerNorm.
// Block = 32 columns (b,l), 8 waves; wave w owns M-rows [32w, 32w+32).
// B tile (32x128 bf16 = 8 KB, contiguous in ybf) is staged into LDS with
// CDNA5 async LDS loads (ASYNCcnt), overlapped with the A-fragment loads.
// ---------------------------------------------------------------------------
__global__ __launch_bounds__(256) void glu_ln_kernel(
    const bf16t* __restrict__ wbf, const bf16t* __restrict__ ybf,
    float* __restrict__ hbuf, const float* __restrict__ out_b,
    const float* __restrict__ ln_g, const float* __restrict__ ln_b,
    int layer) {
  __shared__ float zlds[256 * 33];                    // z tile (256 x 32)
  __shared__ __align__(16) bf16t btile[32 * HDIM];    // B tile (8 KB)
  __shared__ float redsum[8][32];
  __shared__ float redsq[8][32];
  __shared__ float bmu[32], binv[32];

  const int tid = threadIdx.x;
  const int w = tid >> 5;
  const size_t colbase = (size_t)blockIdx.x * 32;
  const bf16t* Yb = ybf + colbase * HDIM;

  // ---- async stage of B tile: 256 lanes x 2 x 16B = 8 KB.  The 24-bit
  // instruction offset is added to BOTH the LDS and global addresses
  // (ISA 15.18.3), so one address pair serves both halves.
  {
    const unsigned lds_off = (unsigned)(uintptr_t)(&btile[0]) + (unsigned)tid * 16u;
    const unsigned long long gaddr =
        (unsigned long long)(uintptr_t)Yb + (unsigned long long)tid * 16ull;
    asm volatile("global_load_async_to_lds_b128 %0, %1, off"
                 :: "v"(lds_off), "v"(gaddr) : "memory");
    asm volatile("global_load_async_to_lds_b128 %0, %1, off offset:4096"
                 :: "v"(lds_off), "v"(gaddr) : "memory");
  }

  // A fragments from global while the async copies are in flight.
  v16bf af[2][4];
#pragma unroll
  for (int mt = 0; mt < 2; ++mt)
#pragma unroll
    for (int kk = 0; kk < 4; ++kk)
      af[mt][kk] = load_a_frag(wbf + (size_t)(w * 32 + mt * 16) * HDIM + kk * 32, HDIM);

  asm volatile("s_wait_asynccnt 0" ::: "memory");
  __syncthreads();

#pragma unroll
  for (int nt = 0; nt < 2; ++nt) {
    v16bf bf[4];
#pragma unroll
    for (int kk = 0; kk < 4; ++kk)
      bf[kk] = load_b_frag(&btile[(nt * 16) * HDIM + kk * 32], HDIM);
#pragma unroll
    for (int mt = 0; mt < 2; ++mt) {
      v8f acc = {};
#pragma unroll
      for (int kk = 0; kk < 4; ++kk) acc = wmma_bf16(af[mt][kk], bf[kk], acc);
      store_d_frag(&zlds[(w * 32 + mt * 16) * 33 + nt * 16], 33, acc);
    }
  }
  __syncthreads();

  // ---- GLU + residual + LayerNorm, 8 threads per column (16 channels each)
  const int j = tid & 31;
  const int q = tid >> 5; // 0..7
  const size_t col = colbase + j;
  const int b = (int)(col >> 12);
  const int l = (int)(col & (LSEQ - 1));
  {
    const float* ob = out_b + layer * 2 * HDIM;
    float sum = 0.f, sumsq = 0.f;
    for (int h2 = q * 16; h2 < q * 16 + 16; ++h2) {
      const float a = zlds[h2 * 33 + j] + ob[h2];
      const float g = zlds[(h2 + HDIM) * 33 + j] + ob[HDIM + h2];
      const float glu = a * (1.f / (1.f + expf(-g)));
      const float res = glu + hbuf[((size_t)b * HDIM + h2) * LSEQ + l];
      zlds[h2 * 33 + j] = res;
      sum += res; sumsq += res * res;
    }
    redsum[q][j] = sum; redsq[q][j] = sumsq;
  }
  __syncthreads();
  if (tid < 32) {
    float s = 0.f, s2 = 0.f;
#pragma unroll
    for (int qq = 0; qq < 8; ++qq) { s += redsum[qq][tid]; s2 += redsq[qq][tid]; }
    const float mu  = s * (1.f / HDIM);
    const float var = s2 * (1.f / HDIM) - mu * mu;
    bmu[tid]  = mu;
    binv[tid] = rsqrtf(var + 1e-5f);
  }
  __syncthreads();
  {
    const float mu = bmu[j], inv = binv[j];
    const float* lg = ln_g + layer * HDIM;
    const float* lb = ln_b + layer * HDIM;
    for (int h2 = q * 16; h2 < q * 16 + 16; ++h2) {
      const float v = (zlds[h2 * 33 + j] - mu) * inv * lg[h2] + lb[h2];
      hbuf[((size_t)b * HDIM + h2) * LSEQ + l] = v;
    }
  }
}

// ---------------------------------------------------------------------------
// Kernel 6: mean-pool over L.  Block per (b,h).
// ---------------------------------------------------------------------------
__global__ __launch_bounds__(256) void pool_kernel(
    const float* __restrict__ hbuf, float* __restrict__ pooled) {
  __shared__ float red[256];
  const int bh = blockIdx.x;
  float s = 0.f;
  for (int l = threadIdx.x; l < LSEQ; l += 256) s += hbuf[(size_t)bh * LSEQ + l];
  red[threadIdx.x] = s;
  __syncthreads();
  for (int st = 128; st > 0; st >>= 1) {
    if (threadIdx.x < st) red[threadIdx.x] += red[threadIdx.x + st];
    __syncthreads();
  }
  if (threadIdx.x == 0) pooled[bh] = red[0] * (1.f / LSEQ);
}

// ---------------------------------------------------------------------------
// Kernel 7: decoder  out(16x100) = pooled(16x128) @ dec_w.T + dec_b
// ---------------------------------------------------------------------------
__global__ __launch_bounds__(256) void decode_kernel(
    const float* __restrict__ pooled, const float* __restrict__ dec_w,
    const float* __restrict__ dec_b, float* __restrict__ out) {
  for (int e = threadIdx.x; e < BATCH * DOUT; e += blockDim.x) {
    const int b = e / DOUT, o = e % DOUT;
    float s = dec_b[o];
    for (int h = 0; h < HDIM; ++h)
      s += pooled[b * HDIM + h] * dec_w[o * HDIM + h];
    out[e] = s;
  }
}

// ---------------------------------------------------------------------------
extern "C" void kernel_launch(void* const* d_in, const int* in_sizes, int n_in,
                              void* d_out, int out_size, void* d_ws, size_t ws_size,
                              hipStream_t stream) {
  const float* x          = (const float*)d_in[0];
  const float* enc_w      = (const float*)d_in[1];
  const float* enc_b      = (const float*)d_in[2];
  const float* log_dt     = (const float*)d_in[3];
  const float* Cp         = (const float*)d_in[4];
  const float* log_A_real = (const float*)d_in[5];
  const float* A_imag     = (const float*)d_in[6];
  const float* Dp         = (const float*)d_in[7];
  const float* out_w      = (const float*)d_in[8];
  const float* out_b      = (const float*)d_in[9];
  const float* ln_g       = (const float*)d_in[10];
  const float* ln_b       = (const float*)d_in[11];
  const float* dec_w      = (const float*)d_in[12];
  const float* dec_b      = (const float*)d_in[13];

  char* ws = (char*)d_ws;
  size_t off = 0;
  auto take = [&](size_t bytes) -> char* {
    char* p = ws + off;
    off += (bytes + 255) & ~(size_t)255;
    return p;
  };
  float* hbuf   = (float*)take(sizeof(float) * BATCH * HDIM * LSEQ);      // 32 MB
  bf16t* ybf    = (bf16t*)take(sizeof(bf16t) * BATCH * LSEQ * HDIM);      // 16 MB
  float* Slocal = (float*)take(sizeof(float) * BATCH * HDIM * 64 * 64);   // 32 MB
  bf16t* Carry  = (bf16t*)take(sizeof(bf16t) * BATCH * HDIM * 64 * 64);   // 16 MB
  bf16t* M1     = (bf16t*)take(sizeof(bf16t) * HDIM * 64 * 64);           // 1 MB
  bf16t* Emat   = (bf16t*)take(sizeof(bf16t) * HDIM * 64 * 64);           // 1 MB
  bf16t* Toep   = (bf16t*)take(sizeof(bf16t) * HDIM * 64 * 64);           // 1 MB
  float* wT     = (float*)take(sizeof(float) * HDIM * NSTATE2 * 2);
  bf16t* wbf    = (bf16t*)take(sizeof(bf16t) * 2 * HDIM * HDIM);
  float* pooled = (float*)take(sizeof(float) * BATCH * HDIM);
  (void)in_sizes; (void)n_in; (void)out_size; (void)ws_size;

  encode_kernel<<<(BATCH * HDIM * LSEQ) / 256, 256, 0, stream>>>(x, enc_w, enc_b, hbuf);

  for (int layer = 0; layer < NLAYER; ++layer) {
    setup_kernel<<<1, 128, 0, stream>>>(log_dt, Cp, log_A_real, A_imag, out_w,
                                        layer, M1, Emat, Toep, wT, wbf);
    phase1_kernel<<<BATCH * HDIM, 128, 0, stream>>>(M1, hbuf, Slocal);
    scan_kernel<<<(BATCH * HDIM * NSTATE2) / 256, 256, 0, stream>>>(Slocal, wT, Carry);
    phase3_kernel<<<BATCH * HDIM, 128, 0, stream>>>(Toep, Emat, Carry, hbuf, Dp, ybf, layer);
    glu_ln_kernel<<<(BATCH * LSEQ) / 32, 256, 0, stream>>>(wbf, ybf, hbuf, out_b, ln_g, ln_b, layer);
  }

  pool_kernel<<<BATCH * HDIM, 256, 0, stream>>>(hbuf, pooled);
  decode_kernel<<<1, 256, 0, stream>>>(pooled, dec_w, dec_b, (float*)d_out);
}
